// Decoder_78606491452142
// MI455X (gfx1250) — matrix operand order repaired and checked by reading, not compiled, not run on hardware
//
#include <hip/hip_runtime.h>
#include <cstdint>
#include <cstddef>

#define V_    32000
#define EMB_  512
#define HID_  512
#define ENC_  512
#define ATT_  256
#define STY_  64
#define B_    32
#define S_    512
#define T_    64
#define XDIM_ 1088   // EMB + STY + ENC
#define G3_   1536   // 3*HID
#define CH_   1024   // ENC + HID

#define BPAD_ 40     // padded k-stride (halves) of transposed B tile: 80B, 16B-aligned

typedef __attribute__((ext_vector_type(16))) __bf16 v16bf;
typedef __attribute__((ext_vector_type(8)))  float  v8f;

__device__ __forceinline__ unsigned short f2us(float f) {
  unsigned u = __float_as_uint(f);
  u += 0x7FFFu + ((u >> 16) & 1u);          // round-to-nearest-even
  return (unsigned short)(u >> 16);
}

// Build a v16bf WMMA operand from two contiguous 16B runs in LDS
// -> two ds_load_b128, no per-element packing.
__device__ __forceinline__ v16bf frag16(const unsigned short* p0,
                                        const unsigned short* p1) {
  union { uint4 q[2]; v16bf b; } c;
  c.q[0] = *(const uint4*)p0;
  c.q[1] = *(const uint4*)p1;
  return c.b;
}

// ---------------- elementwise f32 -> bf16 ----------------
__global__ __launch_bounds__(256) void cvt_bf16(const float* __restrict__ src,
                                                unsigned short* __restrict__ dst,
                                                size_t n) {
  size_t i  = (size_t)blockIdx.x * 256 + threadIdx.x;
  size_t st = (size_t)gridDim.x * 256;
  for (; i < n; i += st) dst[i] = f2us(src[i]);
}

__global__ __launch_bounds__(256) void zero_u32(unsigned* __restrict__ p, size_t n) {
  size_t i = (size_t)blockIdx.x * 256 + threadIdx.x;
  if (i < n) p[i] = 0u;
}

// ---------------- generic big bf16 WMMA GEMM ----------------
// C[M,N] (f32, optional bias) = A[M,K](bf16) * B[K,N](bf16).  TM=64, TN=32.
// Requires M % 64 == 0, N % 32 == 0, K % 32 == 0 (true for all call sites).
// A tile is DMA'd to LDS with global_load_async_to_lds_b128 (ASYNCcnt path);
// B tile is staged transposed so fragments are two ds_load_b128 per operand.
__global__ __launch_bounds__(256) void gemm_big(const unsigned short* __restrict__ A,
                                                const unsigned short* __restrict__ Bm,
                                                float* __restrict__ C,
                                                const float* __restrict__ bias,
                                                int M, int N, int K, int ldc) {
  __shared__ alignas(16) unsigned short sA[64 * 32];       // row-major [m][k]
  __shared__ alignas(16) unsigned short sBT[32 * BPAD_];   // transposed [n][k]
  const int tid = threadIdx.x, lane = tid & 31, wid = tid >> 5;
  const int hl = lane >> 4, r = lane & 15;
  const int m0 = blockIdx.y * 64, n0 = blockIdx.x * 32;
  const int wm = (wid >> 1) * 16, wn = (wid & 1) * 16;
  v8f acc = {};
  const int arow = tid >> 2, ac = (tid & 3) * 8;   // A: 64 rows x 32 halves, 16B/thread
  const int brow = tid >> 3, bc = (tid & 7) * 4;   // B: 32 k-rows, 4 n-halves/thread
  const unsigned sa_dst = (unsigned)(size_t)&sA[arow * 32 + ac];
  const unsigned short* a_src = A + (size_t)(m0 + arow) * K + ac;
  const unsigned short* afp0 = &sA[(wm + r) * 32 + 8 * hl];
  const unsigned short* afp1 = afp0 + 16;
  const unsigned short* bfp0 = &sBT[(wn + r) * BPAD_ + 8 * hl];
  for (int kb = 0; kb < K; kb += 32) {
    // A tile: per-lane 16B async DMA straight into LDS (no VGPR round-trip)
    asm volatile("global_load_async_to_lds_b128 %0, %1, off"
                 :: "v"(sa_dst), "v"(a_src + kb) : "memory");
    // B tile: coalesced row read, transposed scatter into LDS
    {
      uint2 bq = *(const uint2*)(Bm + (size_t)(kb + brow) * N + n0 + bc);
      sBT[(bc + 0) * BPAD_ + brow] = (unsigned short)(bq.x);
      sBT[(bc + 1) * BPAD_ + brow] = (unsigned short)(bq.x >> 16);
      sBT[(bc + 2) * BPAD_ + brow] = (unsigned short)(bq.y);
      sBT[(bc + 3) * BPAD_ + brow] = (unsigned short)(bq.y >> 16);
    }
    if (kb + 32 < K)   // next K-slice of the L2-resident weight panel
      __builtin_prefetch(Bm + (size_t)(kb + 32 + brow) * N + n0 + bc, 0, 1);
    asm volatile("s_wait_asynccnt 0" ::: "memory");
    __syncthreads();
    v16bf af  = frag16(afp0, afp1);          // lane: K = {8hl..8hl+7, 16+8hl..}
    v16bf bfm = frag16(bfp0, bfp0 + 16);     // lane: column wn+r, K = {8hl..8hl+7, 16+8hl..}
    acc = __builtin_amdgcn_wmma_f32_16x16x32_bf16(false, af, false, bfm,
                                                  (short)0, acc, false, false);
    __syncthreads();
  }
  const int col = n0 + wn + r;
  const float bv = bias ? bias[col] : 0.f;
  #pragma unroll
  for (int i = 0; i < 8; ++i) {
    int row = m0 + wm + i + 8 * hl;          // C/D: VGPR i -> M=i (+8 for hi half)
    C[(size_t)row * ldc + col] = acc[i] + bv;
  }
}

// ---------------- per-step GRU gate GEMMs (gi and gh fused in one grid) ------
// TM=32 (=B), TN=64; 8 waves as 2(M) x 4(N).  A composed on the fly from
// [embedding | style | context] (gi) or dec_h (gh), f32->bf16 packed into one
// ds_store_b64 per thread.  B staged transposed as in gemm_big.
__global__ __launch_bounds__(256) void step_gates(
    const float* __restrict__ emb, const int* __restrict__ dec_in,
    const float* __restrict__ sty_e, const int* __restrict__ styles,
    const float* __restrict__ context, const float* __restrict__ dec_h,
    const unsigned short* __restrict__ wih_bf, const unsigned short* __restrict__ whh_bf,
    const float* __restrict__ bih, const float* __restrict__ bhh,
    float* __restrict__ gi, float* __restrict__ gh, int t) {
  __shared__ alignas(16) unsigned short sA[32 * 32];
  __shared__ alignas(16) unsigned short sBT[64 * BPAD_];
  const int tid = threadIdx.x, lane = tid & 31, wid = tid >> 5;
  const int hl = lane >> 4, r = lane & 15;
  const bool isGI = blockIdx.x < 24;
  const int n0 = (isGI ? (int)blockIdx.x : (int)blockIdx.x - 24) * 64;
  const int K  = isGI ? XDIM_ : HID_;
  const unsigned short* Bw = isGI ? wih_bf : whh_bf;
  const float* bias = isGI ? bih : bhh;
  float* out = isGI ? gi : gh;
  const int wm = (wid & 1) * 16, wn = (wid >> 1) * 16;
  v8f acc = {};
  const int arow = tid >> 3, ac = (tid & 7) * 4;    // A: 32 rows x 32 halves
  const int brow = tid >> 3, bc = (tid & 7) * 8;    // B: 32 k-rows x 64 n
  const unsigned short* afp0 = &sA[(wm + r) * 32 + 8 * hl];
  const unsigned short* bfp0 = &sBT[(wn + r) * BPAD_ + 8 * hl];
  for (int kb = 0; kb < K; kb += 32) {
    float4 av;
    const int k0 = kb + ac;
    if (isGI) {
      if (k0 < EMB_) {
        int tok = dec_in[arow * T_ + t];
        av = *(const float4*)(emb + (size_t)tok * EMB_ + k0);
      } else if (k0 < EMB_ + STY_) {
        int st = styles[arow];
        av = *(const float4*)(sty_e + (size_t)st * STY_ + (k0 - EMB_));
      } else {
        av = *(const float4*)(context + (size_t)arow * ENC_ + (k0 - EMB_ - STY_));
      }
    } else {
      av = *(const float4*)(dec_h + (size_t)arow * HID_ + k0);
    }
    {
      uint2 pk;
      pk.x = (unsigned)f2us(av.x) | ((unsigned)f2us(av.y) << 16);
      pk.y = (unsigned)f2us(av.z) | ((unsigned)f2us(av.w) << 16);
      *(uint2*)&sA[arow * 32 + ac] = pk;     // one ds_store_b64
    }
    {
      uint4 bq = *(const uint4*)(Bw + (size_t)(kb + brow) * G3_ + n0 + bc);
      sBT[(bc + 0) * BPAD_ + brow] = (unsigned short)(bq.x);
      sBT[(bc + 1) * BPAD_ + brow] = (unsigned short)(bq.x >> 16);
      sBT[(bc + 2) * BPAD_ + brow] = (unsigned short)(bq.y);
      sBT[(bc + 3) * BPAD_ + brow] = (unsigned short)(bq.y >> 16);
      sBT[(bc + 4) * BPAD_ + brow] = (unsigned short)(bq.z);
      sBT[(bc + 5) * BPAD_ + brow] = (unsigned short)(bq.z >> 16);
      sBT[(bc + 6) * BPAD_ + brow] = (unsigned short)(bq.w);
      sBT[(bc + 7) * BPAD_ + brow] = (unsigned short)(bq.w >> 16);
    }
    if (kb + 32 < K)
      __builtin_prefetch(Bw + (size_t)(kb + 32 + brow) * G3_ + n0 + bc, 0, 1);
    __syncthreads();
    v16bf af  = frag16(afp0, afp0 + 16);
    v16bf bfm = frag16(bfp0, bfp0 + 16);
    acc = __builtin_amdgcn_wmma_f32_16x16x32_bf16(false, af, false, bfm,
                                                  (short)0, acc, false, false);
    __syncthreads();
  }
  const int col = n0 + wn + r;
  const float bv = bias[col];
  #pragma unroll
  for (int i = 0; i < 8; ++i) {
    int row = wm + i + 8 * hl;                        // batch row 0..31 (M == TM)
    out[(size_t)row * G3_ + col] = acc[i] + bv;
  }
}

// ---------------- fused GRU update + coverage attention + context ------------
// One block per batch row.  f32 VALU path — tiny relative to the GEMMs,
// and the nonlinearities want f32.
__global__ __launch_bounds__(256) void attn_step(
    const float* __restrict__ gi, const float* __restrict__ gh,
    float* __restrict__ dec_h, float* __restrict__ context, float* __restrict__ coverage,
    const float* __restrict__ keys_p, const float* __restrict__ enc,
    const int* __restrict__ enc_len,
    const float* __restrict__ wq, const float* __restrict__ bq,
    const float* __restrict__ wcov, const float* __restrict__ wout,
    float* __restrict__ out_attn, int t) {
  __shared__ float sh_h[HID_];
  __shared__ float sh_q[ATT_];
  __shared__ float sh_a[S_];
  __shared__ float red[256];
  const int b = blockIdx.x, tid = threadIdx.x;
  // GRU cell (torch gate order r,z,n)
  for (int j = tid; j < HID_; j += 256) {
    float ir = gi[(size_t)b * G3_ + j];
    float iz = gi[(size_t)b * G3_ + HID_ + j];
    float in_ = gi[(size_t)b * G3_ + 2 * HID_ + j];
    float hr = gh[(size_t)b * G3_ + j];
    float hz = gh[(size_t)b * G3_ + HID_ + j];
    float hn = gh[(size_t)b * G3_ + 2 * HID_ + j];
    float rr = 1.f / (1.f + expf(-(ir + hr)));
    float zz = 1.f / (1.f + expf(-(iz + hz)));
    float nn = tanhf(in_ + rr * hn);
    float h  = (1.f - zz) * nn + zz * dec_h[(size_t)b * HID_ + j];
    dec_h[(size_t)b * HID_ + j] = h;
    sh_h[j] = h;
  }
  __syncthreads();
  // q = dec_h @ wq + bq   (one output per thread)
  {
    float acc = bq[tid];
    for (int e = 0; e < HID_; ++e) acc += sh_h[e] * wq[e * ATT_ + tid];
    sh_q[tid] = acc;
  }
  __syncthreads();
  const int len = enc_len[b];
  // e[s] = tanh(q + keys_p + cov*w_cov) . w_out, masked
  for (int s = tid; s < S_; s += 256) {
    float cov = coverage[(size_t)b * S_ + s];
    const float* kp = keys_p + ((size_t)b * S_ + s) * ATT_;
    float acc = 0.f;
    for (int a = 0; a < ATT_; ++a)
      acc += tanhf(sh_q[a] + kp[a] + cov * wcov[a]) * wout[a];
    sh_a[s] = (s < len) ? acc : -1e10f;
  }
  __syncthreads();
  // softmax over S=512 (2 elements per thread)
  float mx = fmaxf(sh_a[tid], sh_a[tid + 256]);
  red[tid] = mx; __syncthreads();
  for (int st = 128; st > 0; st >>= 1) {
    if (tid < st) red[tid] = fmaxf(red[tid], red[tid + st]);
    __syncthreads();
  }
  mx = red[0]; __syncthreads();
  float e0 = expf(sh_a[tid] - mx), e1 = expf(sh_a[tid + 256] - mx);
  red[tid] = e0 + e1; __syncthreads();
  for (int st = 128; st > 0; st >>= 1) {
    if (tid < st) red[tid] += red[tid + st];
    __syncthreads();
  }
  const float inv = 1.f / red[0];
  float a0 = e0 * inv, a1 = e1 * inv;
  sh_a[tid] = a0; sh_a[tid + 256] = a1;
  coverage[(size_t)b * S_ + tid] += a0;
  coverage[(size_t)b * S_ + tid + 256] += a1;
  out_attn[((size_t)b * T_ + t) * S_ + tid] = a0;
  out_attn[((size_t)b * T_ + t) * S_ + tid + 256] = a1;
  __syncthreads();
  // context = attn @ enc  (2 output dims per thread, coalesced enc reads per s)
  float c0 = 0.f, c1 = 0.f;
  for (int s = 0; s < S_; ++s) {
    float a = sh_a[s];
    const float* er = enc + ((size_t)b * S_ + s) * ENC_;
    c0 += a * er[tid];
    c1 += a * er[tid + 256];
  }
  context[(size_t)b * ENC_ + tid] = c0;
  context[(size_t)b * ENC_ + tid + 256] = c1;
}

// ---------------- h1 = leaky_relu([context|dec_h] @ out_w1 + b1) -> bf16 -----
// Stored at row (b*T + t) so Phase B maps directly onto [B,T,V].
__global__ __launch_bounds__(256) void h1_gemm(
    const float* __restrict__ context, const float* __restrict__ dec_h,
    const unsigned short* __restrict__ w1_bf, const float* __restrict__ b1,
    unsigned short* __restrict__ h1_bf, int t) {
  __shared__ alignas(16) unsigned short sA[32 * 32];
  __shared__ alignas(16) unsigned short sBT[64 * BPAD_];
  const int tid = threadIdx.x, lane = tid & 31, wid = tid >> 5;
  const int hl = lane >> 4, r = lane & 15;
  const int n0 = blockIdx.x * 64;
  const int wm = (wid & 1) * 16, wn = (wid >> 1) * 16;
  v8f acc = {};
  const int arow = tid >> 3, ac = (tid & 7) * 4;
  const int brow = tid >> 3, bc = (tid & 7) * 8;
  const unsigned short* afp0 = &sA[(wm + r) * 32 + 8 * hl];
  const unsigned short* bfp0 = &sBT[(wn + r) * BPAD_ + 8 * hl];
  for (int kb = 0; kb < CH_; kb += 32) {
    const int k0 = kb + ac;
    float4 av = (k0 < ENC_)
        ? *(const float4*)(context + (size_t)arow * ENC_ + k0)
        : *(const float4*)(dec_h + (size_t)arow * HID_ + (k0 - ENC_));
    {
      uint2 pk;
      pk.x = (unsigned)f2us(av.x) | ((unsigned)f2us(av.y) << 16);
      pk.y = (unsigned)f2us(av.z) | ((unsigned)f2us(av.w) << 16);
      *(uint2*)&sA[arow * 32 + ac] = pk;
    }
    {
      uint4 bq = *(const uint4*)(w1_bf + (size_t)(kb + brow) * HID_ + n0 + bc);
      sBT[(bc + 0) * BPAD_ + brow] = (unsigned short)(bq.x);
      sBT[(bc + 1) * BPAD_ + brow] = (unsigned short)(bq.x >> 16);
      sBT[(bc + 2) * BPAD_ + brow] = (unsigned short)(bq.y);
      sBT[(bc + 3) * BPAD_ + brow] = (unsigned short)(bq.y >> 16);
      sBT[(bc + 4) * BPAD_ + brow] = (unsigned short)(bq.z);
      sBT[(bc + 5) * BPAD_ + brow] = (unsigned short)(bq.z >> 16);
      sBT[(bc + 6) * BPAD_ + brow] = (unsigned short)(bq.w);
      sBT[(bc + 7) * BPAD_ + brow] = (unsigned short)(bq.w >> 16);
    }
    __syncthreads();
    v16bf af  = frag16(afp0, afp0 + 16);
    v16bf bfm = frag16(bfp0, bfp0 + 16);
    acc = __builtin_amdgcn_wmma_f32_16x16x32_bf16(false, af, false, bfm,
                                                  (short)0, acc, false, false);
    __syncthreads();
  }
  const int col = n0 + wn + r;
  const float bv = b1[col];
  #pragma unroll
  for (int i = 0; i < 8; ++i) {
    int row = wm + i + 8 * hl;                       // batch b
    float v = acc[i] + bv;
    v = v > 0.f ? v : 0.1f * v;                      // leaky_relu(0.1)
    h1_bf[((size_t)row * T_ + t) * HID_ + col] = f2us(v);
  }
}

// ---------------- log-softmax stats + argmax over V ----------------
__global__ __launch_bounds__(256) void logsm(const float* __restrict__ logits,
                                             const int* __restrict__ dec_out,
                                             float* __restrict__ lp,
                                             float* __restrict__ pred) {
  const int row = blockIdx.x;                        // b*T + t
  const int tid = threadIdx.x;
  const float* lg = logits + (size_t)row * V_;
  __shared__ float rv[256];
  __shared__ int   ri[256];
  __shared__ float rs[256];
  float mx = -3.4e38f; int mi = 0;
  for (int v = tid; v < V_; v += 256) {
    float x = lg[v];
    if (x > mx) { mx = x; mi = v; }
  }
  rv[tid] = mx; ri[tid] = mi; __syncthreads();
  for (int s = 128; s > 0; s >>= 1) {
    if (tid < s) {
      float a = rv[tid], b = rv[tid + s];
      if (b > a || (b == a && ri[tid + s] < ri[tid])) { rv[tid] = b; ri[tid] = ri[tid + s]; }
    }
    __syncthreads();
  }
  const float gmax = rv[0]; const int gidx = ri[0];
  float s = 0.f;
  for (int v = tid; v < V_; v += 256) s += expf(lg[v] - gmax);
  rs[tid] = s; __syncthreads();
  for (int st = 128; st > 0; st >>= 1) {
    if (tid < st) rs[tid] += rs[tid + st];
    __syncthreads();
  }
  if (tid == 0) {
    float lse = gmax + logf(rs[0]);
    int tgt = dec_out[row];
    lp[row] = lg[tgt] - lse;
    pred[row] = (float)gidx;
  }
}

extern "C" void kernel_launch(void* const* d_in, const int* in_sizes, int n_in,
                              void* d_out, int out_size, void* d_ws, size_t ws_size,
                              hipStream_t stream) {
  (void)in_sizes; (void)n_in; (void)out_size; (void)ws_size;
  const float* enc     = (const float*)d_in[0];
  const int*   enc_len = (const int*)d_in[1];
  const int*   styles  = (const int*)d_in[2];
  const int*   dec_in  = (const int*)d_in[3];
  const int*   dec_out = (const int*)d_in[4];
  const float* emb     = (const float*)d_in[5];
  const float* sty_e   = (const float*)d_in[6];
  const float* wq      = (const float*)d_in[7];
  const float* bq      = (const float*)d_in[8];
  const float* wk      = (const float*)d_in[9];
  const float* wcov    = (const float*)d_in[10];
  const float* wout    = (const float*)d_in[11];
  const float* wih     = (const float*)d_in[12];
  const float* whh     = (const float*)d_in[13];
  const float* bih     = (const float*)d_in[14];
  const float* bhh     = (const float*)d_in[15];
  const float* w1      = (const float*)d_in[16];
  const float* b1      = (const float*)d_in[17];
  const float* w2      = (const float*)d_in[18];
  const float* b2      = (const float*)d_in[19];

  char* ws = (char*)d_ws;
  size_t off = 0;
  auto take = [&](size_t bytes) -> void* {
    void* p = ws + off;
    off = (off + bytes + 255) & ~(size_t)255;
    return p;
  };
  unsigned short* enc_bf = (unsigned short*)take((size_t)B_ * S_ * ENC_ * 2);
  unsigned short* wk_bf  = (unsigned short*)take((size_t)ENC_ * ATT_ * 2);
  unsigned short* wih_bf = (unsigned short*)take((size_t)XDIM_ * G3_ * 2);
  unsigned short* whh_bf = (unsigned short*)take((size_t)HID_ * G3_ * 2);
  unsigned short* w1_bf  = (unsigned short*)take((size_t)CH_ * HID_ * 2);
  unsigned short* w2_bf  = (unsigned short*)take((size_t)HID_ * V_ * 2);
  float*          keys_p = (float*)take((size_t)B_ * S_ * ATT_ * 4);
  unsigned short* h1_bf  = (unsigned short*)take((size_t)B_ * T_ * HID_ * 2);
  float*          gi     = (float*)take((size_t)B_ * G3_ * 4);
  float*          gh     = (float*)take((size_t)B_ * G3_ * 4);
  float*          state  = (float*)take((size_t)3 * B_ * HID_ * 4);
  float* dec_h    = state;
  float* context  = state + (size_t)B_ * HID_;
  float* coverage = state + (size_t)2 * B_ * HID_;

  float* out_logits = (float*)d_out;                       // [B,T,V]
  float* out_lp     = out_logits + (size_t)B_ * T_ * V_;   // [B,T]
  float* out_pred   = out_lp + (size_t)B_ * T_;            // [B,T]
  float* out_attn   = out_pred + (size_t)B_ * T_;          // [B,T,S]

  auto cvt = [&](const float* s, unsigned short* d, size_t n) {
    unsigned g = (unsigned)((n + 255) / 256);
    if (g > 8192u) g = 8192u;
    hipLaunchKernelGGL(cvt_bf16, dim3(g), dim3(256), 0, stream, s, d, n);
  };
  cvt(enc, enc_bf, (size_t)B_ * S_ * ENC_);
  cvt(wk,  wk_bf,  (size_t)ENC_ * ATT_);
  cvt(wih, wih_bf, (size_t)XDIM_ * G3_);
  cvt(whh, whh_bf, (size_t)HID_ * G3_);
  cvt(w1,  w1_bf,  (size_t)CH_ * HID_);
  cvt(w2,  w2_bf,  (size_t)HID_ * V_);

  {
    size_t n = (size_t)3 * B_ * HID_;
    hipLaunchKernelGGL(zero_u32, dim3((unsigned)((n + 255) / 256)), dim3(256), 0, stream,
                       (unsigned*)state, n);
  }

  // keys_p = enc_output @ wk  ([16384,512] x [512,256])
  hipLaunchKernelGGL(gemm_big, dim3(ATT_ / 32, (B_ * S_) / 64), dim3(256), 0, stream,
                     enc_bf, wk_bf, keys_p, (const float*)nullptr,
                     B_ * S_, ATT_, ENC_, ATT_);

  // sequential decode: 3 kernels per step, logits deferred to Phase B
  for (int t = 0; t < T_; ++t) {
    hipLaunchKernelGGL(step_gates, dim3(48), dim3(256), 0, stream,
                       emb, dec_in, sty_e, styles, context, dec_h,
                       wih_bf, whh_bf, bih, bhh, gi, gh, t);
    hipLaunchKernelGGL(attn_step, dim3(B_), dim3(256), 0, stream,
                       gi, gh, dec_h, context, coverage, keys_p, enc, enc_len,
                       wq, bq, wcov, wout, out_attn, t);
    hipLaunchKernelGGL(h1_gemm, dim3(HID_ / 64), dim3(256), 0, stream,
                       context, dec_h, w1_bf, b1, h1_bf, t);
  }

  // Phase B: logits = h1 @ out_w2 + b2  ([2048,512] x [512,32000])
  hipLaunchKernelGGL(gemm_big, dim3(V_ / 32, (B_ * T_) / 64), dim3(256), 0, stream,
                     h1_bf, w2_bf, out_logits, b2, B_ * T_, V_, HID_, V_);

  // Phase C: log-softmax stats, target log-prob, argmax
  hipLaunchKernelGGL(logsm, dim3(B_ * T_), dim3(256), 0, stream,
                     out_logits, dec_out, out_lp, out_pred);
}
